// Attention_27230092656979
// MI455X (gfx1250) — compile-verified, matrix operand order checked by reading, not run on hardware
//
#include <hip/hip_runtime.h>

typedef __attribute__((ext_vector_type(4)))  _Float16 v4h;
typedef __attribute__((ext_vector_type(8)))  _Float16 v8h;
typedef __attribute__((ext_vector_type(16))) _Float16 v16h;
typedef __attribute__((ext_vector_type(4)))  float    v4f;
typedef __attribute__((ext_vector_type(8)))  float    v8f;

#define SA_STRIDE  520    // halfs; 64 rows = 66560 B (x tile -> V -> out_heads)
#define SQK_STRIDE 1032   // halfs; 64 rows = 132096 B (Q cols 0..511 | K 512..1023)
#define ATTN_SCALE 0.125f // 64^-0.5

// ---------------- prep: fp32 [K][N] -> f16 transposed [N][K] ----------------
__global__ void cvt_transpose(const float* __restrict__ src,
                              _Float16* __restrict__ dst, int K, int N) {
  int idx = blockIdx.x * blockDim.x + threadIdx.x;
  if (idx < K * N) {
    int k = idx / N, n = idx - k * N;
    dst[(size_t)n * K + k] = (_Float16)src[idx];
  }
}

// 16x32 f16 WMMA fragment (A or B, same K-striping): lane holds row/col =
// lane&15; K chunk base = (lane>>4)*8; elems 0..7 at p, elems 8..15 at p+16.
__device__ __forceinline__ v16h make_frag(const _Float16* p) {
  v8h lo = *(const v8h*)p;
  v8h hi = *(const v8h*)(p + 16);
  v16h f;
#pragma unroll
  for (int t = 0; t < 8; t++) { f[t] = lo[t]; f[t + 8] = hi[t]; }
  return f;
}

// ---------------- fused attention: one block = 8 spatial locations ----------
__global__ __launch_bounds__(256) void attn_fused(
    const float* __restrict__ x, const float* __restrict__ mask,
    const _Float16* __restrict__ wqkvT, const _Float16* __restrict__ woutT,
    const float* __restrict__ b_out, float* __restrict__ out) {
  __shared__ _Float16 sA[64 * SA_STRIDE];   // 66560 B
  __shared__ _Float16 sQK[64 * SQK_STRIDE]; // 132096 B

  const int tid  = threadIdx.x;
  const int lane = tid & 31;
  const int wave = tid >> 5;                // 8 waves
  const int row_base = blockIdx.x * 64;     // 64 flat rows = 8 locations x 8 tokens

  // ---- stage x [64x512] fp32 -> f16 LDS (vectorized, coalesced) ----
  {
    const v4f* xv = (const v4f*)(x + (size_t)row_base * 512);
#pragma unroll
    for (int e = 0; e < 32; e++) {
      int idx4 = e * 256 + tid;          // 0..8191 float4s
      int r = idx4 >> 7, c4 = idx4 & 127;
      v4f f = xv[idx4];
      v4h hh;
#pragma unroll
      for (int t = 0; t < 4; t++) hh[t] = (_Float16)f[t];
      *(v4h*)&sA[r * SA_STRIDE + c4 * 4] = hh;
    }
  }
  __syncthreads();

  const int kb    = (lane >> 4) << 3;  // K sub-base: 0 or 8
  const int nlane = lane & 15;
  const int mbase = (lane >> 4) << 3;  // C-frag row sub-base: 0 or 8

  // ---- GEMM1: [64x512] @ [512x1536] -> qkv; each B frag feeds 4 M-tiles ----
  // 3 passes x 4 N-tiles: pass 0 -> t=0..3, pass 1 -> t=4..7 (all Q,K);
  // pass 2 -> t=8..11 (exactly the V tiles; stored into sA after barrier).
#pragma unroll
  for (int pass = 0; pass < 3; pass++) {
    v8f acc[4][4];
#pragma unroll
    for (int m = 0; m < 4; m++)
#pragma unroll
      for (int u = 0; u < 4; u++) acc[m][u] = (v8f){0, 0, 0, 0, 0, 0, 0, 0};

    for (int kk = 0; kk < 16; ++kk) {
      v16h a[4];
#pragma unroll
      for (int m = 0; m < 4; m++)
        a[m] = make_frag(&sA[(m * 16 + (lane & 15)) * SA_STRIDE + kk * 32 + kb]);
      const _Float16* wbase = wqkvT + kk * 32 + kb + (size_t)nlane * 512;
#pragma unroll
      for (int u = 0; u < 4; u++) {
        int t = pass * 4 + u;
        v16h b = make_frag(wbase + (size_t)(wave + t * 8) * (16 * 512));
#pragma unroll
        for (int m = 0; m < 4; m++)
          acc[m][u] = __builtin_amdgcn_wmma_f32_16x16x32_f16(
              false, a[m], false, b, (short)0, acc[m][u], false, false);
      }
    }
    if (pass < 2) {
      // Q,K tiles -> sQK (disjoint from sA; no barrier needed yet)
#pragma unroll
      for (int u = 0; u < 4; u++) {
        int col = (wave + (pass * 4 + u) * 8) * 16 + nlane;
#pragma unroll
        for (int m = 0; m < 4; m++)
#pragma unroll
          for (int g = 0; g < 8; g++)
            sQK[(m * 16 + mbase + g) * SQK_STRIDE + col] = (_Float16)acc[m][u][g];
      }
    } else {
      __syncthreads();  // all waves done reading sA (x) and writing Q,K
      // V tiles (cols 1024..1535) -> reuse sA
#pragma unroll
      for (int u = 0; u < 4; u++) {
        int col = (wave + (u + 8) * 8) * 16 + nlane - 1024;
#pragma unroll
        for (int m = 0; m < 4; m++)
#pragma unroll
          for (int g = 0; g < 8; g++)
            sA[(m * 16 + mbase + g) * SA_STRIDE + col] = (_Float16)acc[m][u][g];
      }
      __syncthreads();
    }
  }

  // ---- attention: wave w = head h=w for locations l=0..7 ----
  const int h     = wave;
  const int i0    = lane >> 3;        // slot0 rows 0..3 (slot1 = +4)
  const int j0    = lane & 7;
  const int oi    = lane >> 2;        // output row 0..7
  const int dbase = (lane & 3) * 16;  // output d-chunk
  float oacc[8][16];

#pragma unroll
  for (int l = 0; l < 8; l++) {
    const _Float16* q0 = &sQK[(l * 8 + i0)     * SQK_STRIDE + h * 64];
    const _Float16* q1 = &sQK[(l * 8 + i0 + 4) * SQK_STRIDE + h * 64];
    const _Float16* kp = &sQK[(l * 8 + j0)     * SQK_STRIDE + 512 + h * 64];
    float d0 = 0.f, d1 = 0.f;
#pragma unroll
    for (int c = 0; c < 8; c++) {
      v8h kv  = *(const v8h*)(kp + c * 8);
      v8h qv0 = *(const v8h*)(q0 + c * 8);
      v8h qv1 = *(const v8h*)(q1 + c * 8);
#pragma unroll
      for (int t = 0; t < 8; t++) {
        d0 += (float)qv0[t] * (float)kv[t];
        d1 += (float)qv1[t] * (float)kv[t];
      }
    }
    const int loc = blockIdx.x * 8 + l;
    d0 = d0 * ATTN_SCALE * mask[(size_t)loc * 64 + (i0 * 8 + j0)];
    d1 = d1 * ATTN_SCALE * mask[(size_t)loc * 64 + ((i0 + 4) * 8 + j0)];

    // softmax over j (8-lane row groups)
    float m0 = d0, m1 = d1;
#pragma unroll
    for (int mkx = 1; mkx < 8; mkx <<= 1) {
      m0 = fmaxf(m0, __shfl_xor(m0, mkx, 32));
      m1 = fmaxf(m1, __shfl_xor(m1, mkx, 32));
    }
    float e0 = __expf(d0 - m0), e1 = __expf(d1 - m1);
    float s0 = e0, s1 = e1;
#pragma unroll
    for (int mkx = 1; mkx < 8; mkx <<= 1) {
      s0 += __shfl_xor(s0, mkx, 32);
      s1 += __shfl_xor(s1, mkx, 32);
    }
    float a0 = e0 / s0, a1 = e1 / s1;

    // attn @ V: lane owns (row oi, d in [dbase, dbase+16))
#pragma unroll
    for (int t = 0; t < 16; t++) oacc[l][t] = 0.f;
#pragma unroll
    for (int j = 0; j < 8; j++) {
      int src  = (oi * 8 + j) & 31;
      float r0 = __shfl(a0, src, 32);
      float r1 = __shfl(a1, src, 32);
      float aij = (oi < 4) ? r0 : r1;
      const _Float16* vp = &sA[(l * 8 + j) * SA_STRIDE + h * 64 + dbase];
      v8h vlo = *(const v8h*)(vp);
      v8h vhi = *(const v8h*)(vp + 8);
#pragma unroll
      for (int t = 0; t < 8; t++) {
        oacc[l][t]     += aij * (float)vlo[t];
        oacc[l][t + 8] += aij * (float)vhi[t];
      }
    }
  }
  __syncthreads();  // all V reads complete before overwriting with out_heads
#pragma unroll
  for (int l = 0; l < 8; l++) {
    _Float16* op = &sA[(l * 8 + oi) * SA_STRIDE + h * 64 + dbase];
#pragma unroll
    for (int t = 0; t < 16; t++) op[t] = (_Float16)oacc[l][t];
  }
  __syncthreads();

  // ---- GEMM2: [64x512] @ [512x512] + bias -> out (f32) ----
  v8f acc2[4][4];
#pragma unroll
  for (int m = 0; m < 4; m++)
#pragma unroll
    for (int t = 0; t < 4; t++) acc2[m][t] = (v8f){0, 0, 0, 0, 0, 0, 0, 0};
  for (int kk = 0; kk < 16; ++kk) {
    v16h a[4];
#pragma unroll
    for (int m = 0; m < 4; m++)
      a[m] = make_frag(&sA[(m * 16 + (lane & 15)) * SA_STRIDE + kk * 32 + kb]);
    const _Float16* wbase = woutT + kk * 32 + kb + (size_t)nlane * 512;
#pragma unroll
    for (int t = 0; t < 4; t++) {
      v16h b = make_frag(wbase + (size_t)(wave + t * 8) * (16 * 512));
#pragma unroll
      for (int m = 0; m < 4; m++)
        acc2[m][t] = __builtin_amdgcn_wmma_f32_16x16x32_f16(
            false, a[m], false, b, (short)0, acc2[m][t], false, false);
    }
  }
#pragma unroll
  for (int t = 0; t < 4; t++) {
    int col = (wave + t * 8) * 16 + nlane;
    float bias = b_out[col];
#pragma unroll
    for (int m = 0; m < 4; m++)
#pragma unroll
      for (int g = 0; g < 8; g++)
        out[(size_t)(row_base + m * 16 + mbase + g) * 512 + col] =
            acc2[m][t][g] + bias;
  }
}

// ---------------------------------------------------------------------------
extern "C" void kernel_launch(void* const* d_in, const int* in_sizes, int n_in,
                              void* d_out, int out_size, void* d_ws, size_t ws_size,
                              hipStream_t stream) {
  (void)in_sizes; (void)n_in; (void)out_size; (void)ws_size;
  const float* x     = (const float*)d_in[0];
  const float* mask  = (const float*)d_in[1];
  const float* w_qkv = (const float*)d_in[2];  // [512, 1536]
  const float* w_out = (const float*)d_in[3];  // [512, 512]
  const float* b_out = (const float*)d_in[4];  // [512]
  float* out = (float*)d_out;

  _Float16* wqkvT = (_Float16*)d_ws;             // [1536][512] f16
  _Float16* woutT = wqkvT + (size_t)1536 * 512;  // [512][512] f16

  cvt_transpose<<<(512 * 1536 + 255) / 256, 256, 0, stream>>>(w_qkv, wqkvT, 512, 1536);
  cvt_transpose<<<(512 * 512 + 255) / 256, 256, 0, stream>>>(w_out, woutT, 512, 512);

  // 8192 locations / 8 per block = 1024 blocks, 256 threads (8 waves)
  attn_fused<<<1024, 256, 0, stream>>>(x, mask, wqkvT, woutT, b_out, out);
}
